// TextEmbedding_62723702390915
// MI455X (gfx1250) — matrix operand order repaired and checked by reading, not compiled, not run on hardware
//
#include <hip/hip_runtime.h>
#include <hip/hip_bf16.h>

typedef __attribute__((ext_vector_type(16))) _Float16 v16h;
typedef __attribute__((ext_vector_type(8)))  _Float16 v8h;
typedef __attribute__((ext_vector_type(8)))  float    v8f;

#define B_SZ 256
#define T_SZ 512
#define D_SZ 50
#define H_SZ 128      // per-direction hidden
#define G_SZ 512      // 4*H

// workspace layout (bytes)
#define OFF_BWHH  0
#define BWHH_PER_DIR (64*512)                         // uint32 count (k-pairs x 512 cols)
#define OFF_BWIH  (OFF_BWHH + 2*BWHH_PER_DIR*4)       // 262144
#define BWIH_PER_DIR (32*512)                         // uint32 count (K padded to 64)
#define OFF_XG    (OFF_BWIH + 2*BWIH_PER_DIR*4)       // xg fragments, f16 (~256MB)

union AFrag { v16h h; unsigned u[8]; };
union HPack { unsigned u; _Float16 h[2]; };

__device__ __forceinline__ float fsigmoid(float x) { return 1.0f / (1.0f + __expf(-x)); }
__device__ __forceinline__ float ftanh(float x) {
  float xc = fminf(fmaxf(x, -15.0f), 15.0f);
  float e  = __expf(-2.0f * xc);
  return (1.0f - e) / (1.0f + e);
}

// ---------------------------------------------------------------------------
// Pack Whh (transposed, f16 pair-packed along K) and Wih (K zero-padded to 64)
// so WMMA B-fragments become contiguous per-lane dword loads.
// bwhh[dir][k2][n] = {f16 Whh[n][2k2], f16 Whh[n][2k2+1]}   (WhhT pairs)
// bwih[dir][k2][g] = {f16 Wih[g][2k2], f16 Wih[g][2k2+1]}   (zeros for k>=50)
// ---------------------------------------------------------------------------
__global__ void __launch_bounds__(256) prep_pack(
    const float* __restrict__ Whh_f, const float* __restrict__ Whh_b,
    const float* __restrict__ Wih_f, const float* __restrict__ Wih_b,
    unsigned* __restrict__ bwhh, unsigned* __restrict__ bwih) {
  int i = blockIdx.x * blockDim.x + threadIdx.x;   // 0..65535
  {
    int dir = i >> 15;
    int r   = i & 32767;
    int k2  = r >> 9;
    int n   = r & 511;
    const float* W = dir ? Whh_b : Whh_f;
    HPack p;
    p.h[0] = (_Float16)W[n * H_SZ + 2 * k2];
    p.h[1] = (_Float16)W[n * H_SZ + 2 * k2 + 1];
    bwhh[(size_t)dir * BWHH_PER_DIR + k2 * 512 + n] = p.u;
  }
  if (i < 2 * 32 * 512) {
    int dir = i >> 14;
    int r   = i & 16383;
    int k2  = r >> 9;
    int n   = r & 511;
    int k   = 2 * k2;
    const float* W = dir ? Wih_b : Wih_f;
    HPack p;
    p.h[0] = (k     < D_SZ) ? (_Float16)W[n * D_SZ + k]     : (_Float16)0.0f;
    p.h[1] = (k + 1 < D_SZ) ? (_Float16)W[n * D_SZ + k + 1] : (_Float16)0.0f;
    bwih[(size_t)dir * BWIH_PER_DIR + k2 * 512 + n] = p.u;
  }
}

// ---------------------------------------------------------------------------
// Fused embedding gather + input projection:  xg[dir][b,t,4H] = x·Wihᵀ + bias
// Output stored in WMMA C-fragment order (f16, 16B per lane) so the serial
// kernel can C-init each tile with one b128 load.
// Tile id = ((dir*T + t)*16 + btile)*32 + ntile ; elem = tile*256 + lane*8 + r
// ---------------------------------------------------------------------------
__global__ void __launch_bounds__(256) xg_gemm(
    const int*   __restrict__ uidx, const float* __restrict__ emb,
    const float* __restrict__ bih_f, const float* __restrict__ bhh_f,
    const float* __restrict__ bih_b, const float* __restrict__ bhh_b,
    const unsigned* __restrict__ bwih, _Float16* __restrict__ xg) {
  int lane = threadIdx.x & 31;
  int wave = threadIdx.x >> 5;
  int w    = blockIdx.x * 8 + wave;      // 0 .. 16383
  int dir  = w >> 13;
  int rr_  = w & 8191;
  int t    = rr_ >> 4;
  int btile = rr_ & 15;

  int m   = btile * 16 + (lane & 15);    // batch row for A fragment
  int tok = uidx[m * T_SZ + t];
  const float* erow = emb + (size_t)tok * D_SZ;
  int ah = (lane & 16) ? 8 : 0;

  // A fragments, K chunks 0..31 and 32..63 (zero padded past D=50)
  AFrag a0, a1;
#pragma unroll
  for (int j = 0; j < 8; ++j) {
    int koff = (j < 4) ? (ah + 2 * j) : (16 + ah + 2 * (j - 4));
    {
      HPack p;
      p.h[0] = (_Float16)erow[koff];
      p.h[1] = (_Float16)erow[koff + 1];
      a0.u[j] = p.u;
    }
    {
      int k = 32 + koff;
      HPack p;
      if (k < D_SZ) { p.h[0] = (_Float16)erow[k]; p.h[1] = (_Float16)erow[k + 1]; }
      else          { p.u = 0u; }
      a1.u[j] = p.u;
    }
  }

  const unsigned* bw  = bwih + (size_t)dir * BWIH_PER_DIR;
  const float*    bih = dir ? bih_b : bih_f;
  const float*    bhh = dir ? bhh_b : bhh_f;
  int b8   = (lane & 16) ? 4 : 0;
  int ncol = lane & 15;

  for (int nt = 0; nt < 32; ++nt) {
    int n = nt * 16 + ncol;
    AFrag bf0, bf1;
#pragma unroll
    for (int j = 0; j < 8; ++j) {
      bf0.u[j] = bw[(b8 + j + (j & 4)) * 512 + n];
      bf1.u[j] = bw[(16 + b8 + j + (j & 4)) * 512 + n];
    }
    v8f acc = {};
    acc = __builtin_amdgcn_wmma_f32_16x16x32_f16(false, a0.h, false, bf0.h, (short)0, acc, false, false);
    acc = __builtin_amdgcn_wmma_f32_16x16x32_f16(false, a1.h, false, bf1.h, (short)0, acc, false, false);
    float bias = bih[n] + bhh[n];
    v8h outv;
#pragma unroll
    for (int r = 0; r < 8; ++r) outv[r] = (_Float16)(acc[r] + bias);
    size_t tile = ((size_t)(dir * 512 + t) * 16 + btile) * 32 + nt;
    *(v8h*)(xg + tile * 256 + lane * 8) = outv;
  }
}

// ---------------------------------------------------------------------------
// Persistent bidirectional LSTM recurrence + fused masked max-pool.
// Batch-parallel decomposition: 32 independent WGs = 2 dirs x 16 batch tiles.
// No inter-WG communication: per step each WG does a [16x128]x[128x512] GEMM
// for its own 16 batches. Whh B-fragments are register-resident (128 VGPRs);
// h ping-pongs through an 8KB LDS buffer stored in A-FRAGMENT ORDER so each
// lane's A operand is one 32B contiguous ds load; c-state and the masked
// running max never leave VGPRs. One __syncthreads per step is the only sync.
// ---------------------------------------------------------------------------
__global__ void __launch_bounds__(256, 1) bilstm_recurrent(
    const _Float16* __restrict__ xg, const unsigned* __restrict__ bwhh,
    const float* __restrict__ masks, float* __restrict__ out) {
  __shared__ __align__(16) unsigned hA[2][4][32][8];   // [buf][kc][lane][j] dwords

  int lane   = threadIdx.x & 31;
  int kslice = threadIdx.x >> 5;            // wave id = hidden-column slice
  int dir    = blockIdx.x >> 4;
  int btile  = blockIdx.x & 15;

  int ncol = lane & 15;
  int col  = kslice * 16 + ncol;            // hidden column this lane owns (C/D layout)
  int rhi  = (lane >> 4) << 3;              // +8 rows for upper half-wave (C/D layout)
  int b8   = (lane & 16) ? 4 : 0;

  // A-fragment scatter coordinates for writing h[row, col] into LDS fragment
  // order (16-bit A 16x32 lane/VGPR mapping): lane' = row + 16*hi, dword j, half hw
  int kc_w = col >> 5;
  int koff = col & 31;
  int hi16 = (koff & 8) ? 16 : 0;
  int j_w  = ((koff & 16) ? 4 : 0) + ((koff & 7) >> 1);
  int hw_w = koff & 1;

  // Register-resident WhhT fragments: 4 gates x 4 K-chunks x 8 VGPRs = 128 VGPRs
  const unsigned* bw = bwhh + (size_t)dir * BWHH_PER_DIR;
  AFrag bfr[4][4];
#pragma unroll
  for (int gi = 0; gi < 4; ++gi) {
    int n = (gi * 8 + kslice) * 16 + ncol;
#pragma unroll
    for (int kc = 0; kc < 4; ++kc) {
#pragma unroll
      for (int j = 0; j < 8; ++j)
        bfr[gi][kc].u[j] = bw[(kc * 16 + b8 + j + (j & 4)) * 512 + n];
    }
  }

  // zero LDS h buffer 0 (initial hidden state)
  {
    unsigned* z = &hA[0][0][0][0];
    for (int i = threadIdx.x; i < 4 * 32 * 8; i += 256) z[i] = 0u;
  }
  __syncthreads();

  v8f c = {};
  v8f rmax;
#pragma unroll
  for (int r = 0; r < 8; ++r) rmax[r] = -3.0e38f;

  const _Float16* xgbase = xg;
  _Float16* hfrag16 = (_Float16*)&hA[0][0][0][0];

  for (int s = 0; s < T_SZ; ++s) {
    int t   = dir ? (T_SZ - 1 - s) : s;
    int cur = s & 1;
    int nxt = cur ^ 1;

    // C-init: gate preactivations from precomputed xg (bias already folded in)
    v8f acc[4];
#pragma unroll
    for (int gi = 0; gi < 4; ++gi) {
      size_t tile = ((size_t)(dir * 512 + t) * 16 + btile) * 32 + (gi * 8 + kslice);
      v8h xv = *(const v8h*)(xgbase + tile * 256 + lane * 8);
#pragma unroll
      for (int r = 0; r < 8; ++r) acc[gi][r] = (float)xv[r];
    }

    // gates += h · WhhT   (K=128 in 4 chunks of 32); A comes straight out of
    // LDS in fragment order: one 32B contiguous slice per lane per chunk.
#pragma unroll
    for (int kc = 0; kc < 4; ++kc) {
      AFrag af;
      const unsigned* hp = &hA[cur][kc][lane][0];
#pragma unroll
      for (int j = 0; j < 8; ++j) af.u[j] = hp[j];
#pragma unroll
      for (int gi = 0; gi < 4; ++gi)
        acc[gi] = __builtin_amdgcn_wmma_f32_16x16x32_f16(
            false, af.h, false, bfr[gi][kc].h, (short)0, acc[gi], false, false);
    }

    // prefetch next timestep's xg fragments (global_prefetch_b8)
    if (s + 1 < T_SZ) {
      int tn = dir ? (T_SZ - 2 - s) : (s + 1);
#pragma unroll
      for (int gi = 0; gi < 4; ++gi) {
        size_t tile = ((size_t)(dir * 512 + tn) * 16 + btile) * 32 + (gi * 8 + kslice);
        __builtin_prefetch(xgbase + tile * 256 + lane * 8, 0, 1);
      }
    }

    // LSTM cell nonlinearity, h scatter into next LDS fragment buffer,
    // masked running max (all in registers)
    _Float16* hnxt = hfrag16 + (size_t)nxt * (4 * 32 * 8 * 2);
#pragma unroll
    for (int r = 0; r < 8; ++r) {
      float ig = fsigmoid(acc[0][r]);
      float fg = fsigmoid(acc[1][r]);
      float gg = ftanh(acc[2][r]);
      float og = fsigmoid(acc[3][r]);
      float cv = fg * c[r] + ig * gg;
      c[r] = cv;
      float hv = og * ftanh(cv);
      int mrow = r + rhi;                                  // btile-local row 0..15
      int lp   = mrow + hi16;                              // A-layout lane'
      hnxt[((kc_w * 32 + lp) * 8 + j_w) * 2 + hw_w] = (_Float16)hv;
      int brow = btile * 16 + mrow;                        // global batch row
      float pen = (1.0f - masks[brow * T_SZ + t]) * 1.0e8f;
      rmax[r] = fmaxf(rmax[r], hv - pen);
    }

    __syncthreads();
  }

#pragma unroll
  for (int r = 0; r < 8; ++r) {
    int brow = btile * 16 + r + rhi;
    out[(size_t)brow * 256 + dir * 128 + col] = rmax[r];
  }
}

// ---------------------------------------------------------------------------
extern "C" void kernel_launch(void* const* d_in, const int* in_sizes, int n_in,
                              void* d_out, int out_size, void* d_ws, size_t ws_size,
                              hipStream_t stream) {
  const int*   uidx  = (const int*)d_in[0];
  const float* masks = (const float*)d_in[1];
  const float* emb   = (const float*)d_in[2];
  const float* Wih_f = (const float*)d_in[3];
  const float* Whh_f = (const float*)d_in[4];
  const float* bih_f = (const float*)d_in[5];
  const float* bhh_f = (const float*)d_in[6];
  const float* Wih_b = (const float*)d_in[7];
  const float* Whh_b = (const float*)d_in[8];
  const float* bih_b = (const float*)d_in[9];
  const float* bhh_b = (const float*)d_in[10];
  float* out = (float*)d_out;

  char* ws = (char*)d_ws;
  unsigned* bwhh = (unsigned*)(ws + OFF_BWHH);
  unsigned* bwih = (unsigned*)(ws + OFF_BWIH);
  _Float16* xg   = (_Float16*)(ws + OFF_XG);

  prep_pack<<<256, 256, 0, stream>>>(Whh_f, Whh_b, Wih_f, Wih_b, bwhh, bwih);
  xg_gemm<<<2048, 256, 0, stream>>>(uidx, emb, bih_f, bhh_f, bih_b, bhh_b, bwih, xg);
  bilstm_recurrent<<<32, 256, 0, stream>>>(xg, bwhh, masks, out);
}
